// FastAttention_54082228191390
// MI455X (gfx1250) — matrix operand order repaired and checked by reading, not compiled, not run on hardware
//
#include <hip/hip_runtime.h>

// ---------------- CDNA5 WMMA types ----------------
typedef __attribute__((ext_vector_type(16))) __bf16 v16bf;
typedef __attribute__((ext_vector_type(8)))  float  v8f;

union Frag { v16bf v; unsigned u[8]; };

#define WMMA_BF16(a, b, c) \
  __builtin_amdgcn_wmma_f32_16x16x32_bf16(false, (a), false, (b), (short)0, (c), false, false)

// ---------------- problem constants ----------------
constexpr int   Bb     = 4;
constexpr int   Nn     = 16384;
constexpr int   DIMc   = 256;
constexpr int   Hh     = 8;
constexpr int   DHc    = 32;
constexpr int   INNERc = Hh * DHc;     // 256
constexpr int   BN     = Bb * Nn;      // 65536
constexpr float SCALEc = 0.17677669529663687f;  // 32^-0.5
constexpr int   NCHUNK = 64;
constexpr int   CROWS  = Nn / NCHUNK;  // 256

// =====================================================================
// K0: convert weights to bf16 and transpose to [n][k] so B-fragment K
// pairs are contiguous dwords.
// =====================================================================
__global__ __launch_bounds__(256) void prep_weights_kernel(
    const float* __restrict__ Wq, const float* __restrict__ Wk,
    const float* __restrict__ Wv, const float* __restrict__ Wout,
    const float* __restrict__ Wr,
    __bf16* __restrict__ wqT, __bf16* __restrict__ wkT, __bf16* __restrict__ wvT,
    __bf16* __restrict__ woT, __bf16* __restrict__ wrT)
{
  const int i = blockIdx.x * 256 + threadIdx.x;
  if (i < DIMc * INNERc) {
    const int k = i >> 8, nn = i & 255;
    const int o = nn * DIMc + k;
    wqT[o] = (__bf16)Wq[i];
    wkT[o] = (__bf16)Wk[i];
    wvT[o] = (__bf16)Wv[i];
    woT[o] = (__bf16)Wout[i];
  }
  if (i < DHc * DHc) {
    const int k = i >> 5, nn = i & 31;
    wrT[nn * DHc + k] = (__bf16)Wr[i];
  }
}

// =====================================================================
// K1: fused QKV GEMM.  One workgroup = 16 rows.  8 waves each own 6 of
// the 48 16x16 output tiles (q|k|v x 16 col-tiles).  K=256 -> 8 WMMAs
// per tile.  q gets abs(); all outputs stored bf16 (halves HBM traffic).
// =====================================================================
__global__ __launch_bounds__(256) void qkv_gemm_kernel(
    const float* __restrict__ states, const float* __restrict__ agent_qs,
    const float* __restrict__ obs,
    const __bf16* __restrict__ wqT, const __bf16* __restrict__ wkT,
    const __bf16* __restrict__ wvT,
    const float* __restrict__ bq, const float* __restrict__ bk,
    const float* __restrict__ bv,
    __bf16* __restrict__ qb, __bf16* __restrict__ kb, __bf16* __restrict__ vb)
{
  __shared__ __bf16 Ash[3][16][DIMc];
  const int row0 = blockIdx.x * 16;

  for (int i = threadIdx.x; i < 3 * 16 * DIMc; i += 256) {
    const int wch = i / (16 * DIMc);
    const int rem = i - wch * 16 * DIMc;
    const int r = rem >> 8;
    const int c = rem & 255;
    const float* src = (wch == 0) ? states : (wch == 1) ? agent_qs : obs;
    Ash[wch][r][c] = (__bf16)src[(size_t)(row0 + r) * DIMc + c];
  }
  __syncthreads();

  const int lane = threadIdx.x & 31;
  const int wv_  = threadIdx.x >> 5;
  const int half = lane >> 4;
  const int lm   = lane & 15;

  Frag afr[8];
  int cached = -1;
  for (int it = 0; it < 6; ++it) {
    const int t = wv_ * 6 + it;       // 0..47
    const int which = t >> 4;         // 0=q 1=k 2=v
    const int n0 = (t & 15) << 4;
    if (which != cached) {
      #pragma unroll
      for (int ks = 0; ks < 8; ++ks) {
        #pragma unroll
        for (int j = 0; j < 8; ++j) {
          const int kk = ks * 32 + ((j < 4) ? 2 * j : 16 + 2 * (j - 4)) + half * 8;
          afr[ks].u[j] = *reinterpret_cast<const unsigned*>(&Ash[which][lm][kk]);
        }
      }
      cached = which;
    }
    const __bf16* bt  = (which == 0) ? wqT : (which == 1) ? wkT : wvT;
    const float*  bia = (which == 0) ? bq  : (which == 1) ? bk  : bv;
    const int col = n0 + lm;
    const float bval = bia[col];
    v8f acc;
    #pragma unroll
    for (int j = 0; j < 8; ++j) acc[j] = bval;
    const __bf16* brow = bt + (size_t)col * DIMc;
    #pragma unroll
    for (int ks = 0; ks < 8; ++ks) {
      if (ks < 7) __builtin_prefetch(brow + (ks + 1) * 32 + half * 16, 0, 0);
      Frag bf;
      #pragma unroll
      for (int j = 0; j < 8; ++j) {
        const int k = ks * 32 + ((j < 4) ? 2 * j : 16 + 2 * (j - 4)) + half * 8;
        bf.u[j] = *reinterpret_cast<const unsigned*>(brow + k);
      }
      acc = WMMA_BF16(afr[ks].v, bf.v, acc);
    }
    __bf16* outp = (which == 0) ? qb : (which == 1) ? kb : vb;
    #pragma unroll
    for (int j = 0; j < 8; ++j) {
      const int m = j + half * 8;
      float val = acc[j];
      if (which == 0) val = fabsf(val);
      outp[(size_t)(row0 + m) * INNERc + col] = (__bf16)val;
    }
  }
}

// =====================================================================
// K2/K3: chunked online-softmax pooling.  Each block handles one
// (b,h,chunk of 256 rows) -> partial {max, sumexp, acc[32]}.
// =====================================================================
__global__ __launch_bounds__(256) void pool_q_partial_kernel(
    const __bf16* __restrict__ qb, const float* __restrict__ wq_attn,
    float* __restrict__ part)
{
  __shared__ float red[256];
  __shared__ float sacc[DHc];
  const int bh = blockIdx.x / NCHUNK;
  const int chunk = blockIdx.x % NCHUNK;
  const int b = bh / Hh, h = bh % Hh;
  const int n = chunk * CROWS + threadIdx.x;
  const __bf16* row = qb + ((size_t)(b * Nn + n)) * INNERc + h * DHc;
  float qv[DHc];
  float logit = 0.f;
  #pragma unroll
  for (int d = 0; d < DHc; ++d) { qv[d] = (float)row[d]; logit += qv[d] * wq_attn[d]; }
  logit *= SCALEc;
  red[threadIdx.x] = logit; __syncthreads();
  for (int s = 128; s > 0; s >>= 1) {
    if (threadIdx.x < s) red[threadIdx.x] = fmaxf(red[threadIdx.x], red[threadIdx.x + s]);
    __syncthreads();
  }
  const float m = red[0]; __syncthreads();
  const float w = __expf(logit - m);
  red[threadIdx.x] = w;
  if (threadIdx.x < DHc) sacc[threadIdx.x] = 0.f;
  __syncthreads();
  for (int s = 128; s > 0; s >>= 1) {
    if (threadIdx.x < s) red[threadIdx.x] += red[threadIdx.x + s];
    __syncthreads();
  }
  #pragma unroll
  for (int d = 0; d < DHc; ++d) atomicAdd(&sacc[d], w * qv[d]);
  __syncthreads();
  float* p = part + (size_t)blockIdx.x * 36;
  if (threadIdx.x == 0) { p[0] = m; p[1] = red[0]; }
  if (threadIdx.x < DHc) p[2 + threadIdx.x] = sacc[threadIdx.x];
}

__global__ __launch_bounds__(256) void pool_k_partial_kernel(
    const __bf16* __restrict__ kb, const float* __restrict__ wk_attn,
    const float* __restrict__ gq_all, float* __restrict__ part)
{
  __shared__ float red[256];
  __shared__ float sacc[DHc];
  const int bh = blockIdx.x / NCHUNK;
  const int chunk = blockIdx.x % NCHUNK;
  const int b = bh / Hh, h = bh % Hh;
  const int n = chunk * CROWS + threadIdx.x;
  const __bf16* row = kb + ((size_t)(b * Nn + n)) * INNERc + h * DHc;
  const float* gq = gq_all + bh * DHc;
  float kv[DHc];
  float logit = 0.f;
  #pragma unroll
  for (int d = 0; d < DHc; ++d) { kv[d] = (float)row[d]; logit += kv[d] * gq[d] * wk_attn[d]; }
  logit *= SCALEc;
  red[threadIdx.x] = logit; __syncthreads();
  for (int s = 128; s > 0; s >>= 1) {
    if (threadIdx.x < s) red[threadIdx.x] = fmaxf(red[threadIdx.x], red[threadIdx.x + s]);
    __syncthreads();
  }
  const float m = red[0]; __syncthreads();
  const float w = __expf(logit - m);
  red[threadIdx.x] = w;
  if (threadIdx.x < DHc) sacc[threadIdx.x] = 0.f;
  __syncthreads();
  for (int s = 128; s > 0; s >>= 1) {
    if (threadIdx.x < s) red[threadIdx.x] += red[threadIdx.x + s];
    __syncthreads();
  }
  #pragma unroll
  for (int d = 0; d < DHc; ++d) atomicAdd(&sacc[d], w * kv[d]);
  __syncthreads();
  float* p = part + (size_t)blockIdx.x * 36;
  if (threadIdx.x == 0) { p[0] = m; p[1] = red[0]; }
  if (threadIdx.x < DHc) p[2 + threadIdx.x] = sacc[threadIdx.x];
}

// log-sum-exp merge of 64 chunk partials -> global pooled vector [32]
__global__ __launch_bounds__(64) void pool_combine_kernel(
    const float* __restrict__ part, float* __restrict__ gout)
{
  __shared__ float red[64];
  __shared__ float sacc[DHc];
  __shared__ float ssum;
  const int bh = blockIdx.x;
  const float* p = part + (size_t)(bh * NCHUNK + threadIdx.x) * 36;
  const float m = p[0];
  red[threadIdx.x] = m; __syncthreads();
  for (int s = 32; s > 0; s >>= 1) {
    if (threadIdx.x < s) red[threadIdx.x] = fmaxf(red[threadIdx.x], red[threadIdx.x + s]);
    __syncthreads();
  }
  const float M = red[0]; __syncthreads();
  const float sc = __expf(m - M);
  if (threadIdx.x == 0) ssum = 0.f;
  if (threadIdx.x < DHc) sacc[threadIdx.x] = 0.f;
  __syncthreads();
  atomicAdd(&ssum, p[1] * sc);
  #pragma unroll
  for (int d = 0; d < DHc; ++d) atomicAdd(&sacc[d], p[2 + d] * sc);
  __syncthreads();
  if (threadIdx.x < DHc) gout[bh * DHc + threadIdx.x] = sacc[threadIdx.x] / ssum;
}

// =====================================================================
// K5: fused epilogue.  The q tile is a pure bf16 copy -> route it over
// the CDNA5 async path (GLOBAL_LOAD_ASYNC_TO_LDS_B128, ASYNCcnt) so it
// overlaps the VALU work building u = v * global_k.  Then:
// r = u@Wr + br + q (per-head K=32 WMMA) -> bf16 requantize in LDS ->
// out = r@Wout + bout.
// =====================================================================
__global__ __launch_bounds__(256) void final_kernel(
    const __bf16* __restrict__ qb, const __bf16* __restrict__ vb,
    const float* __restrict__ gk, const __bf16* __restrict__ wrT,
    const float* __restrict__ br, const __bf16* __restrict__ woT,
    const float* __restrict__ bout, float* __restrict__ out)
{
  __shared__ __bf16 Ush[16][INNERc];
  __shared__ __bf16 Qsh[16][INNERc];
  __shared__ __bf16 Rsh[16][INNERc];
  const int row0 = blockIdx.x * 16;
  const int b = row0 / Nn;     // 16-row tile never crosses batch boundary
  const float* gkb = gk + b * Hh * DHc;

  // ---- async copy: qb tile (8KB bf16) -> Qsh via GLOBAL_LOAD_ASYNC_TO_LDS_B128
  {
    const char* qsrc = (const char*)(qb + (size_t)row0 * INNERc);
    const unsigned ldsbase = (unsigned)(uintptr_t)(void*)&Qsh[0][0];
    #pragma unroll
    for (int it = 0; it < 2; ++it) {
      const unsigned off = (unsigned)(it * 256 + threadIdx.x) * 16u;
      const unsigned lds = ldsbase + off;
      const unsigned long long ga = (unsigned long long)(uintptr_t)(qsrc + off);
      asm volatile("global_load_async_to_lds_b128 %0, %1, off"
                   :: "v"(lds), "v"(ga) : "memory");
    }
  }

  // ---- overlapped VALU work: u = v * global_k (needs lanes anyway)
  for (int i = threadIdx.x; i < 16 * INNERc; i += 256) {
    const int r = i >> 8, c = i & 255;
    const size_t idx = (size_t)(row0 + r) * INNERc + c;
    Ush[r][c] = (__bf16)((float)vb[idx] * gkb[c]);  // c == h*32+d
  }
  asm volatile("s_wait_asynccnt 0x0" ::: "memory");
  __syncthreads();

  const int lane = threadIdx.x & 31;
  const int wv_  = threadIdx.x >> 5;
  const int half = lane >> 4;
  const int lm   = lane & 15;

  // stage 1: per-head r = u @ Wr + br + q   (16 tiles, 2 per wave, K=32)
  #pragma unroll
  for (int it = 0; it < 2; ++it) {
    const int t = wv_ * 2 + it;           // 0..15
    const int h = t >> 1;
    const int dcol = (t & 1) * 16 + lm;   // column within head
    const int col = h * DHc + dcol;
    Frag a, bf;
    #pragma unroll
    for (int j = 0; j < 8; ++j) {
      const int kk = h * DHc + ((j < 4) ? 2 * j : 16 + 2 * (j - 4)) + half * 8;
      a.u[j] = *reinterpret_cast<const unsigned*>(&Ush[lm][kk]);
    }
    #pragma unroll
    for (int j = 0; j < 8; ++j) {
      const int k = ((j < 4) ? 2 * j : 16 + 2 * (j - 4)) + half * 8;
      bf.u[j] = *reinterpret_cast<const unsigned*>(wrT + (size_t)dcol * DHc + k);
    }
    const float bval = br[dcol];
    v8f acc;
    #pragma unroll
    for (int j = 0; j < 8; ++j) acc[j] = bval;
    acc = WMMA_BF16(a.v, bf.v, acc);
    #pragma unroll
    for (int j = 0; j < 8; ++j) {
      const int m = j + half * 8;
      Rsh[m][col] = (__bf16)(acc[j] + (float)Qsh[m][col]);
    }
  }
  __syncthreads();

  // stage 2: out = r @ Wout + bout   (16 col tiles, 2 per wave, K=256)
  #pragma unroll
  for (int it = 0; it < 2; ++it) {
    const int t = wv_ * 2 + it;
    const int col = t * 16 + lm;
    const float bval = bout[col];
    v8f acc;
    #pragma unroll
    for (int j = 0; j < 8; ++j) acc[j] = bval;
    const __bf16* brow = woT + (size_t)col * DIMc;
    #pragma unroll
    for (int ks = 0; ks < 8; ++ks) {
      if (ks < 7) __builtin_prefetch(brow + (ks + 1) * 32 + half * 16, 0, 0);
      Frag a, bf;
      #pragma unroll
      for (int j = 0; j < 8; ++j) {
        const int kk = ks * 32 + ((j < 4) ? 2 * j : 16 + 2 * (j - 4)) + half * 8;
        a.u[j] = *reinterpret_cast<const unsigned*>(&Rsh[lm][kk]);
      }
      #pragma unroll
      for (int j = 0; j < 8; ++j) {
        const int k = ks * 32 + ((j < 4) ? 2 * j : 16 + 2 * (j - 4)) + half * 8;
        bf.u[j] = *reinterpret_cast<const unsigned*>(brow + k);
      }
      acc = WMMA_BF16(a.v, bf.v, acc);
    }
    #pragma unroll
    for (int j = 0; j < 8; ++j) {
      out[(size_t)(row0 + j + half * 8) * DIMc + col] = acc[j];
    }
  }
}

// =====================================================================
// host-side orchestration (graph-capture safe: only kernel launches)
// =====================================================================
extern "C" void kernel_launch(void* const* d_in, const int* in_sizes, int n_in,
                              void* d_out, int out_size, void* d_ws, size_t ws_size,
                              hipStream_t stream) {
  (void)in_sizes; (void)n_in; (void)out_size; (void)ws_size;
  const float* states   = (const float*)d_in[0];
  const float* agent_qs = (const float*)d_in[1];
  const float* obs      = (const float*)d_in[2];
  const float* Wq       = (const float*)d_in[3];
  const float* bq       = (const float*)d_in[4];
  const float* Wk       = (const float*)d_in[5];
  const float* bk       = (const float*)d_in[6];
  const float* Wv       = (const float*)d_in[7];
  const float* bv       = (const float*)d_in[8];
  const float* wq_attn  = (const float*)d_in[9];
  const float* wk_attn  = (const float*)d_in[10];
  const float* Wr       = (const float*)d_in[11];
  const float* br       = (const float*)d_in[12];
  const float* Wout     = (const float*)d_in[13];
  const float* bout     = (const float*)d_in[14];
  float* out = (float*)d_out;

  // workspace carve-up (~102 MB): bf16 q/k/v + bf16 transposed weights + partials
  __bf16* qb  = (__bf16*)d_ws;
  __bf16* kb  = qb  + (size_t)BN * INNERc;
  __bf16* vb  = kb  + (size_t)BN * INNERc;
  __bf16* wqT = vb  + (size_t)BN * INNERc;
  __bf16* wkT = wqT + DIMc * INNERc;
  __bf16* wvT = wkT + DIMc * INNERc;
  __bf16* woT = wvT + DIMc * INNERc;
  __bf16* wrT = woT + DIMc * INNERc;
  float* qpart = (float*)(wrT + DHc * DHc);
  float* kpart = qpart + (size_t)Bb * Hh * NCHUNK * 36;
  float* gq    = kpart + (size_t)Bb * Hh * NCHUNK * 36;
  float* gk    = gq + Bb * Hh * DHc;

  prep_weights_kernel<<<DIMc * INNERc / 256, 256, 0, stream>>>(
      Wq, Wk, Wv, Wout, Wr, wqT, wkT, wvT, woT, wrT);
  qkv_gemm_kernel<<<BN / 16, 256, 0, stream>>>(
      states, agent_qs, obs, wqT, wkT, wvT, bq, bk, bv, qb, kb, vb);
  pool_q_partial_kernel<<<Bb * Hh * NCHUNK, 256, 0, stream>>>(qb, wq_attn, qpart);
  pool_combine_kernel<<<Bb * Hh, 64, 0, stream>>>(qpart, gq);
  pool_k_partial_kernel<<<Bb * Hh * NCHUNK, 256, 0, stream>>>(kb, wk_attn, gq, kpart);
  pool_combine_kernel<<<Bb * Hh, 64, 0, stream>>>(kpart, gk);
  final_kernel<<<BN / 16, 256, 0, stream>>>(qb, vb, gk, wrT, br, woT, bout, out);
}
